// GroupedQueryAttention_37134287241952
// MI455X (gfx1250) — matrix operand order repaired and checked by reading, not compiled, not run on hardware
//
#include <hip/hip_runtime.h>
#include <hip/hip_bf16.h>
#include <hip/hip_fp16.h>

typedef __attribute__((ext_vector_type(16))) _Float16 v16h;
typedef __attribute__((ext_vector_type(8)))  _Float16 v8h;
typedef __attribute__((ext_vector_type(8)))  float    v8f;

#define DIM_C   2048
#define T_SEQ   2048
#define B_SZ    2
#define NH      16
#define NKV     4
#define HD      128
#define MTOT    (B_SZ * T_SEQ)   // 4096
#define KVDIM   (NKV * HD)       // 512

// ---------------------------------------------------------------------------
// Fragment loaders, matching CDNA5 WMMA VGPR layouts (CDNA5 ISA).
// A (16x32 f16): lanes 0-15 -> row M=lane, K={0..7,16..23}; lanes 16-31 -> K={8..15,24..31}
// B (32x16 f16): lane -> col N=lane%16, K same as A: lanes 0-15 K={0..7,16..23}; lanes 16-31 K={8..15,24..31}
// Both loaders assume the K dimension is contiguous in memory (NT layout).
// ---------------------------------------------------------------------------
__device__ __forceinline__ v16h load_a_frag(const _Float16* A, int lda, int row0, int k0) {
    int lane = threadIdx.x & 31;
    const _Float16* p = A + (size_t)(row0 + (lane & 15)) * lda + k0 + ((lane >> 4) << 3);
    v8h lo = *(const v8h*)(p);
    v8h hi = *(const v8h*)(p + 16);
    v16h a;
#pragma unroll
    for (int i = 0; i < 8; ++i) { a[i] = lo[i]; a[8 + i] = hi[i]; }
    return a;
}

__device__ __forceinline__ v16h load_b_frag(const _Float16* Bt, int ldb, int col0, int k0) {
    int lane = threadIdx.x & 31;
    const _Float16* p = Bt + (size_t)(col0 + (lane & 15)) * ldb + k0 + ((lane >> 4) << 3);
    v8h lo = *(const v8h*)(p);
    v8h hi = *(const v8h*)(p + 16);
    v16h b;
#pragma unroll
    for (int i = 0; i < 8; ++i) { b[i] = lo[i]; b[8 + i] = hi[i]; }
    return b;
}

// ---------------------------------------------------------------------------
// Cast helpers
// ---------------------------------------------------------------------------
__global__ void cast_f32_f16(const float* __restrict__ in, _Float16* __restrict__ out, size_t n) {
    size_t i = (size_t)blockIdx.x * blockDim.x + threadIdx.x;
    size_t stride = (size_t)gridDim.x * blockDim.x;
    for (; i < n; i += stride) out[i] = (_Float16)in[i];
}

// in: [rows][cols] f32 -> out: [cols][rows] f16 (transpose so K becomes contiguous)
__global__ void cast_f32_f16_T(const float* __restrict__ in, _Float16* __restrict__ out,
                               int rows, int cols) {
    size_t n = (size_t)rows * cols;
    size_t i = (size_t)blockIdx.x * blockDim.x + threadIdx.x;
    size_t stride = (size_t)gridDim.x * blockDim.x;
    for (; i < n; i += stride) {
        int r = (int)(i / cols);
        int c = (int)(i - (size_t)r * cols);
        out[(size_t)c * rows + r] = (_Float16)in[i];
    }
}

// ---------------------------------------------------------------------------
// NT WMMA GEMM: C[M,N] = A[M,K] (row-major f16) * Bt[N,K]^T (f16, K-contiguous)
// mode 0: f16 row-major C (ldc=N)
// mode 1: f32 row-major C (ldc=N)  -> final projection into d_out
// mode 2: f16 "d-major" store for V: C laid out per-batch as [N][T_SEQ]
// One wave computes a 32x64 tile (2 A-frags x 4 B-frags, 8 WMMA / k-step),
// with double-buffered fragment loads so loads overlap the matrix pipe.
// 4 waves / block.
// ---------------------------------------------------------------------------
__global__ void __launch_bounds__(128)
gemm_nt_f16(const _Float16* __restrict__ A, const _Float16* __restrict__ Bt,
            void* __restrict__ C, int M, int N, int K, int mode) {
    int wave = threadIdx.x >> 5;
    int lane = threadIdx.x & 31;
    int m0 = (blockIdx.x * 4 + wave) * 32;
    int n0 = blockIdx.y * 64;
    if (m0 >= M || n0 >= N) return;

    v8f zero = {};
    v8f acc[2][4];
#pragma unroll
    for (int mi = 0; mi < 2; ++mi)
#pragma unroll
        for (int nc = 0; nc < 4; ++nc) acc[mi][nc] = zero;

    // Prologue: stage k=0 fragments
    v16h a_cur[2], b_cur[4];
    a_cur[0] = load_a_frag(A, K, m0,      0);
    a_cur[1] = load_a_frag(A, K, m0 + 16, 0);
#pragma unroll
    for (int nc = 0; nc < 4; ++nc) b_cur[nc] = load_b_frag(Bt, K, n0 + nc * 16, 0);

    int k0 = 0;
    for (; k0 < K - 32; k0 += 32) {
        // Issue next-step loads first so they fly under the WMMAs below.
        v16h a_nxt[2], b_nxt[4];
        a_nxt[0] = load_a_frag(A, K, m0,      k0 + 32);
        a_nxt[1] = load_a_frag(A, K, m0 + 16, k0 + 32);
#pragma unroll
        for (int nc = 0; nc < 4; ++nc) b_nxt[nc] = load_b_frag(Bt, K, n0 + nc * 16, k0 + 32);
        __builtin_prefetch(A + (size_t)(m0 + (lane & 15)) * K + k0 + 128, 0, 1);

#pragma unroll
        for (int nc = 0; nc < 4; ++nc) {
            acc[0][nc] = __builtin_amdgcn_wmma_f32_16x16x32_f16(
                false, a_cur[0], false, b_cur[nc], (short)0, acc[0][nc], false, false);
            acc[1][nc] = __builtin_amdgcn_wmma_f32_16x16x32_f16(
                false, a_cur[1], false, b_cur[nc], (short)0, acc[1][nc], false, false);
        }
        a_cur[0] = a_nxt[0]; a_cur[1] = a_nxt[1];
#pragma unroll
        for (int nc = 0; nc < 4; ++nc) b_cur[nc] = b_nxt[nc];
    }
    // Epilogue k-step
#pragma unroll
    for (int nc = 0; nc < 4; ++nc) {
        acc[0][nc] = __builtin_amdgcn_wmma_f32_16x16x32_f16(
            false, a_cur[0], false, b_cur[nc], (short)0, acc[0][nc], false, false);
        acc[1][nc] = __builtin_amdgcn_wmma_f32_16x16x32_f16(
            false, a_cur[1], false, b_cur[nc], (short)0, acc[1][nc], false, false);
    }

    int rowoff = (lane >> 4) << 3;   // 0 for lanes 0-15, 8 for lanes 16-31
    int coloff = lane & 15;
#pragma unroll
    for (int mi = 0; mi < 2; ++mi) {
#pragma unroll
        for (int nc = 0; nc < 4; ++nc) {
#pragma unroll
            for (int r = 0; r < 8; ++r) {
                int m = m0 + mi * 16 + rowoff + r;
                int n = n0 + nc * 16 + coloff;
                float v = acc[mi][nc][r];
                if (mode == 0) {
                    ((_Float16*)C)[(size_t)m * N + n] = (_Float16)v;
                } else if (mode == 1) {
                    ((float*)C)[(size_t)m * N + n] = v;
                } else {
                    int bb = m / T_SEQ;
                    int t  = m - bb * T_SEQ;
                    ((_Float16*)C)[(size_t)bb * N * T_SEQ + (size_t)n * T_SEQ + t] = (_Float16)v;
                }
            }
        }
    }
}

// ---------------------------------------------------------------------------
// Flash-style grouped-query attention.
// grid = (T_SEQ/64, B_SZ*NH), 128 threads (4 waves). Each wave: 16 Q rows of one
// (batch, head). Q: [MTOT][DIM] f16, K: [MTOT][KVDIM] f16 (row-major),
// V: d-major per (b,kv): [B][KVDIM][T_SEQ] f16. Out: [MTOT][DIM] f16.
// 1/sqrt(d) is folded into the Q fragments. K fragments are loaded as a batch
// before the S WMMAs; V fragments are loaded before the softmax VALU chain so
// the exp/shuffle work hides their latency.
// ---------------------------------------------------------------------------
__global__ void __launch_bounds__(128)
gqa_flash(const _Float16* __restrict__ Qh, const _Float16* __restrict__ Kh,
          const _Float16* __restrict__ Vt, _Float16* __restrict__ AOh) {
    __shared__ _Float16 pshare[4][16 * 32];

    int wave = threadIdx.x >> 5;
    int lane = threadIdx.x & 31;
    int bh = blockIdx.y;
    int b  = bh >> 4;
    int h  = bh & 15;
    int kv = h >> 2;               // NH/NKV = 4 query heads per kv head
    int row0 = (blockIdx.x * 4 + wave) * 16;

    const _Float16* Qbase = Qh + (size_t)(b * T_SEQ + row0) * DIM_C + h * HD;
    const _Float16* Kbase = Kh + (size_t)b * T_SEQ * KVDIM + kv * HD;
    const _Float16* Vbase = Vt + (size_t)b * KVDIM * T_SEQ + (size_t)kv * HD * T_SEQ;

    const float scale = 0.08838834764831845f;   // 1/sqrt(HD)
    v16h qf[4];
#pragma unroll
    for (int c = 0; c < 4; ++c) {
        qf[c] = load_a_frag(Qbase, DIM_C, 0, c * 32);
#pragma unroll
        for (int i = 0; i < 16; ++i) qf[c][i] = (_Float16)((float)qf[c][i] * scale);
    }

    v8f zero = {};
    v8f o[8];
#pragma unroll
    for (int i = 0; i < 8; ++i) o[i] = zero;
    float mrow[8], lrow[8];
#pragma unroll
    for (int r = 0; r < 8; ++r) { mrow[r] = -1e30f; lrow[r] = 0.0f; }

    _Float16* lds = pshare[wave];
    int rowoff = (lane >> 4) << 3;
    int coloff = lane & 15;

    for (int j0 = 0; j0 < T_SEQ; j0 += 32) {
        // Batch-load all 8 K fragments, then run the 8 S WMMAs.
        v16h kf[8];
#pragma unroll
        for (int c = 0; c < 4; ++c) {
            kf[c]     = load_b_frag(Kbase, KVDIM, j0,      c * 32);
            kf[4 + c] = load_b_frag(Kbase, KVDIM, j0 + 16, c * 32);
        }
        v8f s0 = zero, s1 = zero;
#pragma unroll
        for (int c = 0; c < 4; ++c) {
            s0 = __builtin_amdgcn_wmma_f32_16x16x32_f16(false, qf[c], false, kf[c],     (short)0, s0, false, false);
            s1 = __builtin_amdgcn_wmma_f32_16x16x32_f16(false, qf[c], false, kf[4 + c], (short)0, s1, false, false);
        }

        // V fragments are independent of the softmax: issue their loads now so
        // the exp/shuffle chain below hides the latency.
        v16h vf[8];
#pragma unroll
        for (int nc = 0; nc < 8; ++nc) vf[nc] = load_b_frag(Vbase, T_SEQ, nc * 16, j0);

        // Online softmax over this key block (row reductions across 16 lanes)
        float corr[8];
#pragma unroll
        for (int r = 0; r < 8; ++r) {
            float a0 = s0[r];
            float a1 = s1[r];
            float mx = fmaxf(a0, a1);
#pragma unroll
            for (int off = 1; off < 16; off <<= 1)
                mx = fmaxf(mx, __shfl_xor(mx, off, 32));
            float mnew = fmaxf(mrow[r], mx);
            float c0 = __expf(mrow[r] - mnew);
            float p0 = __expf(a0 - mnew);
            float p1 = __expf(a1 - mnew);
            float sum = p0 + p1;
#pragma unroll
            for (int off = 1; off < 16; off <<= 1)
                sum += __shfl_xor(sum, off, 32);
            lrow[r] = lrow[r] * c0 + sum;
            mrow[r] = mnew;
            corr[r] = c0;
            int row = r + rowoff;
            lds[row * 32 + coloff]      = (_Float16)p0;
            lds[row * 32 + 16 + coloff] = (_Float16)p1;
        }
#pragma unroll
        for (int nc = 0; nc < 8; ++nc)
#pragma unroll
            for (int r = 0; r < 8; ++r) o[nc][r] *= corr[r];

        __syncthreads();
        v16h pf = load_a_frag(lds, 32, 0, 0);   // P (16x32) in A-fragment layout
        __syncthreads();

        // O += P * V_block
#pragma unroll
        for (int nc = 0; nc < 8; ++nc) {
            o[nc] = __builtin_amdgcn_wmma_f32_16x16x32_f16(false, pf, false, vf[nc], (short)0, o[nc], false, false);
        }
    }

#pragma unroll
    for (int r = 0; r < 8; ++r) lrow[r] = 1.0f / lrow[r];
#pragma unroll
    for (int nc = 0; nc < 8; ++nc)
#pragma unroll
        for (int r = 0; r < 8; ++r) {
            int row = row0 + rowoff + r;
            int col = h * HD + nc * 16 + coloff;
            AOh[(size_t)(b * T_SEQ + row) * DIM_C + col] = (_Float16)(o[nc][r] * lrow[r]);
        }
}

// ---------------------------------------------------------------------------
// Launcher
// ---------------------------------------------------------------------------
extern "C" void kernel_launch(void* const* d_in, const int* in_sizes, int n_in,
                              void* d_out, int out_size, void* d_ws, size_t ws_size,
                              hipStream_t stream) {
    const float* x  = (const float*)d_in[0];
    const float* Wq = (const float*)d_in[1];
    const float* Wk = (const float*)d_in[2];
    const float* Wv = (const float*)d_in[3];
    const float* Wo = (const float*)d_in[4];
    float* out = (float*)d_out;

    _Float16* ws = (_Float16*)d_ws;
    size_t off = 0;
    _Float16* xh  = ws + off; off += (size_t)MTOT * DIM_C;     // 8M
    _Float16* Wqt = ws + off; off += (size_t)DIM_C * DIM_C;    // 4M
    _Float16* Wkt = ws + off; off += (size_t)KVDIM * DIM_C;    // 1M
    _Float16* Wvt = ws + off; off += (size_t)KVDIM * DIM_C;    // 1M
    _Float16* Wot = ws + off; off += (size_t)DIM_C * DIM_C;    // 4M
    _Float16* Qh  = ws + off; off += (size_t)MTOT * DIM_C;     // 8M
    _Float16* Kh  = ws + off; off += (size_t)MTOT * KVDIM;     // 2M
    _Float16* Vt  = ws + off; off += (size_t)B_SZ * KVDIM * T_SEQ; // 2M
    _Float16* AOh = ws + off; off += (size_t)MTOT * DIM_C;     // 8M

    // Stage f16 operands (weights transposed so WMMA B loads are contiguous)
    cast_f32_f16  <<<1024, 256, 0, stream>>>(x, xh, (size_t)MTOT * DIM_C);
    cast_f32_f16_T<<<1024, 256, 0, stream>>>(Wq, Wqt, DIM_C, DIM_C);
    cast_f32_f16_T<<<1024, 256, 0, stream>>>(Wk, Wkt, DIM_C, KVDIM);
    cast_f32_f16_T<<<1024, 256, 0, stream>>>(Wv, Wvt, DIM_C, KVDIM);
    cast_f32_f16_T<<<1024, 256, 0, stream>>>(Wo, Wot, DIM_C, DIM_C);

    // Projections (each block covers 4 waves * 32 rows = 128 rows of M)
    dim3 gq(MTOT / 128, DIM_C / 64);
    dim3 gk(MTOT / 128, KVDIM / 64);
    gemm_nt_f16<<<gq, 128, 0, stream>>>(xh, Wqt, Qh, MTOT, DIM_C, DIM_C, 0);
    gemm_nt_f16<<<gk, 128, 0, stream>>>(xh, Wkt, Kh, MTOT, KVDIM, DIM_C, 0);
    gemm_nt_f16<<<gk, 128, 0, stream>>>(xh, Wvt, Vt, MTOT, KVDIM, DIM_C, 2); // d-major V

    // Attention
    dim3 ga(T_SEQ / 64, B_SZ * NH);
    gqa_flash<<<ga, 128, 0, stream>>>(Qh, Kh, Vt, AOh);

    // Output projection -> f32 d_out
    gemm_nt_f16<<<gq, 128, 0, stream>>>(AOh, Wot, out, MTOT, DIM_C, DIM_C, 1);
}
